// DiscreteContinuousEncoder_27968827031820
// MI455X (gfx1250) — matrix operand order repaired and checked by reading, not compiled, not run on hardware
//
#include <hip/hip_runtime.h>

typedef float  v2f   __attribute__((ext_vector_type(2)));
typedef float  v8f   __attribute__((ext_vector_type(8)));
typedef __bf16 v16bf __attribute__((ext_vector_type(16)));
typedef unsigned int v4u __attribute__((ext_vector_type(4)));

// Problem constants (match reference)
constexpr int CIN  = 26;
constexpr int HIN  = 721;
constexpr int WIN  = 1440;
constexpr int NK   = 9;            // K "kernels"
constexpr int HOUT = 360;
constexpr int KH   = 9;
constexpr int KW   = 9;
constexpr int COUT = 256;
constexpr int WOUT = 720;

constexpr int CK   = CIN * NK;     // 234: GEMM reduction dim
constexpr int KP2  = 256;          // stage-2 K padded to 8 x 32
constexpr int ASTR = 264;          // accT row stride (ushorts): 528B -> conflict-friendly
constexpr int WT   = 16;           // w tile (GEMM N)
constexpr int NWT  = WOUT / WT;    // 45
constexpr int XC   = 40;           // x tile columns (2*15 + 8 + pad), 20 b64 chunks
constexpr int TKP  = 84;           // stage-1 taps 81 padded to 21 x 4

union Pack { v4u q[2]; v16bf v; }; // two 16B chunks <-> one bf16x16 WMMA operand

__device__ __forceinline__ void split_bf16(float x, unsigned short& h, unsigned short& l) {
    // hi = truncate-to-bf16(x); lo = truncate-to-bf16(x - hi).  hi+lo ~ fp32 to ~2^-16 rel.
    unsigned xb = __float_as_uint(x);
    unsigned hb = xb & 0xffff0000u;
    float hfv = __uint_as_float(hb);
    float lfv = x - hfv;
    h = (unsigned short)(hb >> 16);
    l = (unsigned short)(__float_as_uint(lfv) >> 16);
}

// ---- prep: split weight[256][234] f32 -> bf16 hi/lo, K zero-padded to 256 ----
__global__ __launch_bounds__(256)
void split_weight_kernel(const float* __restrict__ w, unsigned short* __restrict__ wsp)
{
    const int i = blockIdx.x * 256 + threadIdx.x;
    if (i >= COUT * KP2) return;
    const int o = i >> 8;
    const int k = i & (KP2 - 1);
    const float v = (k < CK) ? w[o * CK + k] : 0.0f;
    unsigned short hh, ll;
    split_bf16(v, hh, ll);
    wsp[i] = hh;                  // A-hi  [COUT][KP2]
    wsp[COUT * KP2 + i] = ll;     // A-lo  [COUT][KP2]
}

// ---- fused: per (h, 16-w tile): f32-WMMA stencil -> bf16x3-WMMA GEMM ----
__global__ __launch_bounds__(256)
void disco_encoder_fused(const float* __restrict__ x,
                         const float* __restrict__ psi,
                         const int*   __restrict__ hi_base,
                         const unsigned short* __restrict__ wsp,
                         float*       __restrict__ out)
{
    __shared__ float xs[CIN * KH * XC];            // 37,440 B: x tile (fp32, exact)
    __shared__ float psiA[16 * TKP];               //  5,376 B: psi as stage-1 A (rows 9..15 zero)
    __shared__ unsigned short accTh[WT * ASTR];    //  8,448 B: stage-2 B hi  [w][ck]
    __shared__ unsigned short accTl[WT * ASTR];    //  8,448 B: stage-2 B lo  [w][ck]

    const int tid = threadIdx.x;
    const int h   = blockIdx.x / NWT;
    const int w0  = (blockIdx.x % NWT) * WT;
    const int hb  = hi_base[h];                    // uniform -> scalar load

    // ---- x tile via async global->LDS copies (ASYNCcnt, no VGPR staging) ----
    // Pairs of floats: start col is always even -> never straddles the mod-1440
    // seam, and both global & LDS addresses are 8B aligned.
    {
        const int NP = CIN * KH * (XC / 2);        // 4680 b64 chunks
        for (int i = tid; i < NP; i += 256) {
            const int c  = i / (KH * (XC / 2));
            const int r  = i - c * (KH * (XC / 2));
            const int dh = r / (XC / 2);
            const int jp = r - dh * (XC / 2);      // pair index 0..19
            int row = hb + dh;
            row = row < 0 ? 0 : (row > HIN - 1 ? HIN - 1 : row);
            int col = 2 * w0 - 4 + 2 * jp + WIN;   // even, in [1436, 2882]
            if (col >= WIN) col -= WIN;
            if (col >= WIN) col -= WIN;
            const float* gsrc = x + ((size_t)(c * HIN + row) * WIN + col);
            // wave-relative LDS byte offset = low 32 bits of the generic pointer
            unsigned ldsOff = (unsigned)(size_t)(&xs[c * (KH * XC) + dh * XC + 2 * jp]);
            asm volatile("global_load_async_to_lds_b64 %0, %1, off"
                         :: "v"(ldsOff), "v"(gsrc)
                         : "memory");
        }
    }

    // psi -> stage-1 A matrix [k=0..15][t=0..83], zero outside 9x81 (overlaps async)
    for (int i = tid; i < 16 * TKP; i += 256) {
        const int k = i / TKP;
        const int t = i - k * TKP;
        psiA[i] = (k < NK && t < KH * KW) ? psi[(k * HOUT + h) * (KH * KW) + t] : 0.0f;
    }
    // zero stage-2 K padding (ck = 234..263) so the GEMM tail is exact
    for (int i = tid; i < WT * (ASTR - CK); i += 256) {
        const int w  = i / (ASTR - CK);
        const int ck = CK + i - w * (ASTR - CK);
        accTh[w * ASTR + ck] = 0;
        accTl[w * ASTR + ck] = 0;
    }

    asm volatile("s_wait_asynccnt 0x0" ::: "memory");  // this wave's LDS copies done
    __syncthreads();

    const int lane = tid & 31;
    const int wave = tid >> 5;
    const int hf   = lane >> 4;                    // lane half
    const int lm   = lane & 15;

    // ---- stage 1: per-channel GEMM acc[k,w] = sum_t psi[k,t] * X[c,t,w] -----
    // exact fp32 on V_WMMA_F32_16X16X4_F32; A (psi) hoisted to registers.
    v2f a1[21];
    int bo0[21], bo1[21];
    #pragma unroll
    for (int ks = 0; ks < 21; ++ks) {
        const int t2 = 4 * ks + 2 * hf;            // A: lane half owns K = {t2, t2+1}
        a1[ks] = *(const v2f*)&psiA[lm * TKP + t2];
        const int ta = t2     > 80 ? 80 : t2;      // t>=81: A is zero, clamp B addr in-bounds
        const int tb = t2 + 1 > 80 ? 80 : t2 + 1;
        bo0[ks] = (ta / KW) * XC + 2 * lm + (ta - (ta / KW) * KW);
        bo1[ks] = (tb / KW) * XC + 2 * lm + (tb - (tb / KW) * KW);
    }

    for (int c = wave; c < CIN; c += 8) {          // wave-uniform -> EXEC all-ones at WMMA
        const float* xb = &xs[c * (KH * XC)];
        v8f d = {};
        #pragma unroll
        for (int ks = 0; ks < 21; ++ks) {
            v2f b;
            b.x = xb[bo0[ks]];
            b.y = xb[bo1[ks]];
            d = __builtin_amdgcn_wmma_f32_16x16x4_f32(
                    false, a1[ks], false, b, (short)0, d, false, false);
        }
        // D rows k=0..8 are valid -> split to bf16 hi/lo stage-2 B rows ck=9c+k
        const int ckb = c * NK;
        if (hf == 0) {
            #pragma unroll
            for (int i = 0; i < 8; ++i) {          // k = 0..7
                unsigned short hh, ll;
                split_bf16(d[i], hh, ll);
                accTh[lm * ASTR + ckb + i] = hh;
                accTl[lm * ASTR + ckb + i] = ll;
            }
        } else {                                   // k = 8 only (k=9..15 belong to next c)
            unsigned short hh, ll;
            split_bf16(d[0], hh, ll);
            accTh[lm * ASTR + ckb + 8] = hh;
            accTl[lm * ASTR + ckb + 8] = ll;
        }
    }
    __syncthreads();

    // ---- stage 2: out[256x16] = W[256x234] * acc[234x16], bf16x3 WMMA -------
    const unsigned short* wh = wsp;
    const unsigned short* wl = wsp + COUT * KP2;
    const unsigned short* brh = &accTh[lm * ASTR]; // B: lane owns column N = lm
    const unsigned short* brl = &accTl[lm * ASTR];

    #pragma unroll
    for (int mt = 0; mt < 2; ++mt) {
        const int moBase = (wave + 8 * mt) * 16;
        const unsigned short* whr = wh + (size_t)(moBase + lm) * KP2;  // A row M = lm
        const unsigned short* wlr = wl + (size_t)(moBase + lm) * KP2;

        v8f acc = {};
        #pragma unroll
        for (int ks = 0; ks < 8; ++ks) {
            const int k0 = 32 * ks;
            // A bf16 layout: lane half owns K chunks {k0+8hf..+7} and {k0+16+8hf..+7}
            Pack ah, al;
            ah.q[0] = *(const v4u*)(whr + k0 + 8 * hf);
            ah.q[1] = *(const v4u*)(whr + k0 + 16 + 8 * hf);
            al.q[0] = *(const v4u*)(wlr + k0 + 8 * hf);
            al.q[1] = *(const v4u*)(wlr + k0 + 16 + 8 * hf);
            // B bf16 layout: lane half owns K chunks {k0+8hf..+7} and {k0+16+8hf..+7}
            Pack bh, bl;
            bh.q[0] = *(const v4u*)(brh + k0 + 8 * hf);
            bh.q[1] = *(const v4u*)(brh + k0 + 16 + 8 * hf);
            bl.q[0] = *(const v4u*)(brl + k0 + 8 * hf);
            bl.q[1] = *(const v4u*)(brl + k0 + 16 + 8 * hf);
            // bf16x3: AhBh + AhBl + AlBh (AlBl negligible), f32 accumulate
            acc = __builtin_amdgcn_wmma_f32_16x16x32_bf16(
                      false, ah.v, false, bh.v, (short)0, acc, false, false);
            acc = __builtin_amdgcn_wmma_f32_16x16x32_bf16(
                      false, ah.v, false, bl.v, (short)0, acc, false, false);
            acc = __builtin_amdgcn_wmma_f32_16x16x32_bf16(
                      false, al.v, false, bh.v, (short)0, acc, false, false);
        }
        // store D: vgpr i -> M = moBase + 8*hf + i, N = lm
        float* ob = out + (size_t)h * WOUT + w0 + lm;
        #pragma unroll
        for (int i = 0; i < 8; ++i) {
            const int o = moBase + 8 * hf + i;
            ob[(size_t)o * (HOUT * WOUT)] = acc[i];
        }
    }
}

extern "C" void kernel_launch(void* const* d_in, const int* in_sizes, int n_in,
                              void* d_out, int out_size, void* d_ws, size_t ws_size,
                              hipStream_t stream) {
    const float* x       = (const float*)d_in[0];
    const float* psi     = (const float*)d_in[1];
    const float* weight  = (const float*)d_in[2];
    const int*   hi_base = (const int*)d_in[3];
    float*       out     = (float*)d_out;
    unsigned short* wsp  = (unsigned short*)d_ws;  // needs 2*256*256*2 = 256 KB scratch

    // 1) split weight into bf16 hi/lo planes, zero-padded to K=256
    split_weight_kernel<<<(COUT * KP2 + 255) / 256, 256, 0, stream>>>(weight, wsp);
    // 2) fused stencil (f32 WMMA) + GEMM (bf16x3 WMMA), async x-tile staging
    disco_encoder_fused<<<HOUT * NWT, 256, 0, stream>>>(x, psi, hi_base, wsp, out);
}
